// PerformerAttentionWrapper_30133490549483
// MI455X (gfx1250) — matrix operand order repaired and checked by reading, not compiled, not run on hardware
//
#include <hip/hip_runtime.h>
#include <math.h>

// Performer (FAVOR+) attention block for MI455X / gfx1250, wave32, bf16 WMMA.
// B=4, N=4096, H=16, DH=64, D=1024, M=266 (padded to MP=288 = 9*32).

#define B_   4
#define N_   4096
#define H_   16
#define DH_  64
#define D_   1024
#define M_   266
#define MP_  288

#define DN_         0.35355339059327373f   // 64^-0.25
#define DIAG_SCALE_ 0.0625f                // 0.5 * DN_^2
#define RATIO_      0.06131393076047564f   // 266^-0.5
#define EPS_FAVOR_  1e-4f
#define EPS_LN_     1e-5f

typedef __attribute__((ext_vector_type(16))) __bf16 v16bf;
typedef __attribute__((ext_vector_type(8)))  float  v8f;

// ---------------------------------------------------------------------------
// Helpers
// ---------------------------------------------------------------------------
__device__ __forceinline__ v8f wmma_bf16(v16bf a, v16bf b, v8f c) {
  return __builtin_amdgcn_wmma_f32_16x16x32_bf16(false, a, false, b,
                                                 (short)0, c, false, false);
}

// Pack two f32 -> packed 2x bf16 dword (lowers to v_cvt_pk_bf16_f32).
__device__ __forceinline__ unsigned pack2(float a, float b) {
  const unsigned short lo = __builtin_bit_cast(unsigned short, (__bf16)a);
  const unsigned short hi = __builtin_bit_cast(unsigned short, (__bf16)b);
  return (unsigned)lo | ((unsigned)hi << 16);
}

// A fragment (16x32 bf16): lane L -> row L%16; elems 0..7 = K (L/16)*8 .. +7,
// elems 8..15 = K 16+(L/16)*8 .. +7 (ISA 7.12.2 16-bit A layout).
__device__ __forceinline__ v16bf frag_a(const __bf16* As, int row0, int ld) {
  const int lane = threadIdx.x & 31;
  const __bf16* p = As + (size_t)(row0 + (lane & 15)) * ld;
  const int k0 = (lane >> 4) << 3;
  v16bf a;
#pragma unroll
  for (int j = 0; j < 8; ++j) {
    a[j]     = p[k0 + j];
    a[j + 8] = p[16 + k0 + j];
  }
  return a;
}

// B fragment (32x16 bf16): lane L -> column L%16; elems 0..7 = K (L/16)*8 .. +7,
// elems 8..15 = K 16+(L/16)*8 .. +7.
__device__ __forceinline__ v16bf frag_b(const __bf16* Bs, int col0, int ld) {
  const int lane = threadIdx.x & 31;
  const __bf16* p = Bs + col0 + (lane & 15);
  const int k0 = (lane >> 4) << 3;
  v16bf b;
#pragma unroll
  for (int j = 0; j < 8; ++j) {
    b[j]     = p[(size_t)(k0 + j) * ld];
    b[j + 8] = p[(size_t)(16 + k0 + j) * ld];
  }
  return b;
}

// C/D 16x16 f32: VGPR r -> row r + 8*(lane/16), col = lane%16.
__device__ __forceinline__ void store_tile(float* C, int ldc, int row0, int col0,
                                           v8f acc, float alpha,
                                           const float* rowScale,
                                           int rowLim, int colLim) {
  const int lane = threadIdx.x & 31;
  const int col = col0 + (lane & 15);
  if (col >= colLim) return;
  const int rbase = row0 + ((lane >> 4) << 3);
#pragma unroll
  for (int r = 0; r < 8; ++r) {
    const int row = rbase + r;
    if (row < rowLim) {
      float v = acc[r] * alpha;
      if (rowScale) v *= rowScale[row];
      C[(size_t)row * ldc + col] = v;
    }
  }
}

__device__ __forceinline__ float wave_sum(float v) {
#pragma unroll
  for (int o = 16; o > 0; o >>= 1) v += __shfl_xor(v, o, 32);
  return v;
}
__device__ __forceinline__ float wave_max(float v) {
#pragma unroll
  for (int o = 16; o > 0; o >>= 1) v = fmaxf(v, __shfl_xor(v, o, 32));
  return v;
}

// ---------------------------------------------------------------------------
// K1/K10: C[16384x1024] = A[16384x1024] @ W[1024x1024]  (QKV + out proj)
// Block tile 128x128, 8 waves as 4x2; each wave 32x64 (2x4 WMMA accums).
// B staged as two 64-wide LDS panels (stride 72 bf16, conflict-friendly).
// ---------------------------------------------------------------------------
__global__ __launch_bounds__(256)
void gemm_xw_kernel(const float* __restrict__ A, const float* __restrict__ W,
                    float* __restrict__ C) {
  __shared__ unsigned AsW[128 * 20];      // 128 x 32 bf16, stride 40
  __shared__ unsigned BsW[2][32 * 36];    // 2 panels of 32 x 64 bf16, stride 72
  const int tid = threadIdx.x;
  const int w = tid >> 5, wr = w >> 1, wc = w & 1;
  const int row0 = blockIdx.x * 128;
  const int col0 = blockIdx.y * 128;

  // Per-thread fixed staging slots (uniform trip, constant strides).
  const int ar  = tid >> 4;               // 0..15  (+16 per pass, 8 passes)
  const int ac2 = (tid & 15) * 2;         // 0..30
  const float* ag = A + (size_t)(row0 + ar) * D_ + ac2;
  unsigned* asp = AsW + ar * 20 + (ac2 >> 1);

  const int br  = tid >> 6;               // 0..3   (+4 per pass, 8 passes)
  const int bc2 = (tid & 63) * 2;         // 0..126
  const int bsel = bc2 >> 6;
  const int blc  = bc2 & 63;
  const float* wg = W + (size_t)br * D_ + col0 + bc2;
  unsigned* bsp = BsW[bsel] + br * 36 + (blc >> 1);

  v8f acc[2][4] = {};

  for (int k0 = 0; k0 < D_; k0 += 32) {
#pragma unroll
    for (int p = 0; p < 8; ++p) {
      const float2 v = *(const float2*)(ag + (size_t)p * 16 * D_ + k0);
      asp[p * 16 * 20] = pack2(v.x, v.y);
    }
#pragma unroll
    for (int p = 0; p < 8; ++p) {
      const float2 v = *(const float2*)(wg + (size_t)(k0 + p * 4) * D_);
      bsp[p * 4 * 36] = pack2(v.x, v.y);
    }
    if (k0 + 32 < D_) {  // global_prefetch_b8 of the next K tile
      __builtin_prefetch(ag + k0 + 32, 0, 1);
      __builtin_prefetch(wg + (size_t)(k0 + 32) * D_, 0, 1);
    }
    __syncthreads();
    const __bf16* As = (const __bf16*)AsW;
    const __bf16* Bp = (const __bf16*)BsW[wc];
    const v16bf a0 = frag_a(As, wr * 32,      40);
    const v16bf a1 = frag_a(As, wr * 32 + 16, 40);
#pragma unroll
    for (int j = 0; j < 4; ++j) {
      const v16bf bj = frag_b(Bp, j * 16, 72);
      acc[0][j] = wmma_bf16(a0, bj, acc[0][j]);
      acc[1][j] = wmma_bf16(a1, bj, acc[1][j]);
    }
    __syncthreads();
  }
#pragma unroll
  for (int j = 0; j < 4; ++j) {
    store_tile(C, D_, row0 + wr * 32,      col0 + wc * 64 + j * 16, acc[0][j], 1.f, nullptr, 1 << 30, 1 << 30);
    store_tile(C, D_, row0 + wr * 32 + 16, col0 + wc * 64 + j * 16, acc[1][j], 1.f, nullptr, 1 << 30, 1 << 30);
  }
}

// ---------------------------------------------------------------------------
// K2: xd[b,h][n,m] = DN * X_head[n,:64] . proj[m,:64]   (proj^T staged in LDS)
// Block tile 64x64, 8 waves as 4x2; each wave 16x32 (1x2 accums). K = 64.
// ---------------------------------------------------------------------------
__global__ __launch_bounds__(256)
void feature_gemm_kernel(const float* __restrict__ X, const float* __restrict__ proj,
                         float* __restrict__ Cd) {
  __shared__ unsigned AsW[64 * 20];       // 64 x 32 bf16, stride 40
  __shared__ __bf16  Bs[32 * 72];         // 32 x 64 bf16, stride 72
  const int tid = threadIdx.x;
  const int w = tid >> 5, wr = w >> 1, wc = w & 1;
  const int bh = blockIdx.z, b = bh >> 4, h = bh & 15;
  const int n0 = blockIdx.x * 64;
  const int m0 = blockIdx.y * 64;
  const float* Ab = X + (size_t)b * N_ * D_ + (size_t)h * DH_;
  float* Cb = Cd + (size_t)bh * N_ * MP_;

  const int ar  = tid >> 4;               // 0..15  (+16 per pass, 4 passes)
  const int ac2 = (tid & 15) * 2;
  const float* ag = Ab + (size_t)(n0 + ar) * D_ + ac2;
  unsigned* asp = AsW + ar * 20 + (ac2 >> 1);

  const int pm  = tid >> 4;               // proj row within tile (+16, 4 passes)
  const int pk2 = (tid & 15) * 2;         // contiguous k pair

  v8f acc0 = {}, acc1 = {};

#pragma unroll
  for (int k0 = 0; k0 < DH_; k0 += 32) {
#pragma unroll
    for (int p = 0; p < 4; ++p) {
      const float2 v = *(const float2*)(ag + (size_t)p * 16 * D_ + k0);
      asp[p * 16 * 20] = pack2(v.x, v.y);
    }
#pragma unroll
    for (int p = 0; p < 4; ++p) {         // transpose proj -> Bs[k][m]
      const int m = pm + p * 16;
      float2 v = {0.f, 0.f};
      if (m0 + m < M_) v = *(const float2*)(proj + (size_t)(m0 + m) * DH_ + k0 + pk2);
      Bs[pk2 * 72 + m]       = (__bf16)v.x;
      Bs[(pk2 + 1) * 72 + m] = (__bf16)v.y;
    }
    __syncthreads();
    const __bf16* As = (const __bf16*)AsW;
    const v16bf a  = frag_a(As, wr * 16, 40);
    const v16bf b0 = frag_b(Bs, wc * 32,      72);
    const v16bf b1 = frag_b(Bs, wc * 32 + 16, 72);
    acc0 = wmma_bf16(a, b0, acc0);
    acc1 = wmma_bf16(a, b1, acc1);
    __syncthreads();
  }
  store_tile(Cb, MP_, n0 + wr * 16, m0 + wc * 32,      acc0, DN_, nullptr, N_, MP_);
  store_tile(Cb, MP_, n0 + wr * 16, m0 + wc * 32 + 16, acc1, DN_, nullptr, N_, MP_);
}

// ---------------------------------------------------------------------------
// K3: query features, in place: qf = RATIO*(exp(xd - diag - rowmax)+eps)
// ---------------------------------------------------------------------------
__global__ __launch_bounds__(256)
void qexp_kernel(float* __restrict__ Fd, const float* __restrict__ Xq) {
  const int lane = threadIdx.x & 31, wid = threadIdx.x >> 5;
  const int bh = blockIdx.y, b = bh >> 4, h = bh & 15;
  const int n = blockIdx.x * 8 + wid;
  const float* xr = Xq + (size_t)b * N_ * D_ + (size_t)n * D_ + (size_t)h * DH_;
  float s = xr[lane] * xr[lane] + xr[lane + 32] * xr[lane + 32];
  s = wave_sum(s);
  const float diag = DIAG_SCALE_ * s;
  float* fr = Fd + (size_t)bh * N_ * MP_ + (size_t)n * MP_;
  float mx = -1e30f;
  for (int m = lane; m < M_; m += 32) mx = fmaxf(mx, fr[m]);
  mx = wave_max(mx);
  for (int m = lane; m < MP_; m += 32) {
    float v = 0.f;
    if (m < M_) v = RATIO_ * (expf(fr[m] - diag - mx) + EPS_FAVOR_);
    fr[m] = v;
  }
}

// K4: global max over kd per (b,h). Row-wise, no integer division.
__global__ __launch_bounds__(1024)
void kmax_kernel(const float* __restrict__ Kd, float* __restrict__ kmax) {
  __shared__ float sm[32];
  const int bh = blockIdx.x;
  const int lane = threadIdx.x & 31, wid = threadIdx.x >> 5;
  const float* base = Kd + (size_t)bh * N_ * MP_;
  float mx = -1e30f;
  for (int n = wid; n < N_; n += 32) {
    const float* r = base + (size_t)n * MP_;
    for (int m = lane; m < M_; m += 32) mx = fmaxf(mx, r[m]);
  }
  mx = wave_max(mx);
  if (lane == 0) sm[wid] = mx;
  __syncthreads();
  if (threadIdx.x < 32) {
    float v = wave_max(sm[threadIdx.x]);
    if (threadIdx.x == 0) kmax[bh] = v;
  }
}

// K5: key features, stab = global max
__global__ __launch_bounds__(256)
void kexp_kernel(float* __restrict__ Fd, const float* __restrict__ Xk,
                 const float* __restrict__ kmax) {
  const int lane = threadIdx.x & 31, wid = threadIdx.x >> 5;
  const int bh = blockIdx.y, b = bh >> 4, h = bh & 15;
  const int n = blockIdx.x * 8 + wid;
  const float* xr = Xk + (size_t)b * N_ * D_ + (size_t)n * D_ + (size_t)h * DH_;
  float s = xr[lane] * xr[lane] + xr[lane + 32] * xr[lane + 32];
  s = wave_sum(s);
  const float diag = DIAG_SCALE_ * s;
  const float stab = kmax[bh];
  float* fr = Fd + (size_t)bh * N_ * MP_ + (size_t)n * MP_;
  for (int m = lane; m < MP_; m += 32) {
    float v = 0.f;
    if (m < M_) v = RATIO_ * (expf(fr[m] - diag - stab) + EPS_FAVOR_);
    fr[m] = v;
  }
}

// K6: k_sum[b,h,m] = sum_n kf[b,h,n,m]
__global__ void ksum_kernel(const float* __restrict__ Kf, float* __restrict__ ksum) {
  const int bh = blockIdx.x, m = threadIdx.x;  // MP_ threads
  const float* base = Kf + (size_t)bh * N_ * MP_ + m;
  float s = 0.f;
  for (int n = 0; n < N_; ++n) s += base[(size_t)n * MP_];
  ksum[(size_t)bh * MP_ + m] = s;
}

// ---------------------------------------------------------------------------
// K7: ctx[b,h][m,d] = sum_n kf[n,m] * v[n,d]   (kf^T staged into LDS)
// Block tile 64(m) x 64(d), K = N_ = 4096 in steps of 32.
// ---------------------------------------------------------------------------
__global__ __launch_bounds__(256)
void ctx_gemm_kernel(const float* __restrict__ Kf, const float* __restrict__ V,
                     float* __restrict__ Ctx) {
  __shared__ __bf16   As[64 * 40];        // kf^T tile: 64(m) x 32(n)
  __shared__ unsigned BsW[32 * 36];       // V tile: 32 x 64 bf16, stride 72
  const int tid = threadIdx.x;
  const int w = tid >> 5, wr = w >> 1, wc = w & 1;
  const int bh = blockIdx.z, b = bh >> 4, h = bh & 15;
  const int m0 = blockIdx.x * 64;
  const float* kfb = Kf + (size_t)bh * N_ * MP_;
  const float* Vb  = V + (size_t)b * N_ * D_ + (size_t)h * DH_;
  float* Cb = Ctx + (size_t)bh * MP_ * DH_;

  const int kk0 = tid >> 5;               // 0..7   (+8 per pass, 4 passes)
  const int mm2 = (tid & 31) * 2;         // 0..62
  const bool mok = (m0 + mm2) < MP_;

  const int vr  = tid >> 5;               // 0..7   (+8 per pass, 4 passes)
  const int vc2 = (tid & 31) * 2;
  const float* vg = Vb + (size_t)vr * D_ + vc2;
  unsigned* bsp = BsW + vr * 36 + (vc2 >> 1);

  v8f acc0 = {}, acc1 = {};

  for (int n0 = 0; n0 < N_; n0 += 32) {
#pragma unroll
    for (int p = 0; p < 4; ++p) {         // transpose kf -> As[m][n]
      const int kk = kk0 + p * 8;
      float2 v = {0.f, 0.f};
      if (mok) v = *(const float2*)(kfb + (size_t)(n0 + kk) * MP_ + m0 + mm2);
      As[mm2 * 40 + kk]       = (__bf16)v.x;
      As[(mm2 + 1) * 40 + kk] = (__bf16)v.y;
    }
#pragma unroll
    for (int p = 0; p < 4; ++p) {
      const float2 v = *(const float2*)(vg + (size_t)(n0 + p * 8) * D_);
      bsp[p * 8 * 36] = pack2(v.x, v.y);
    }
    __syncthreads();
    const __bf16* Bp = (const __bf16*)BsW;
    const v16bf a  = frag_a(As, wr * 16, 40);
    const v16bf b0 = frag_b(Bp, wc * 32,      72);
    const v16bf b1 = frag_b(Bp, wc * 32 + 16, 72);
    acc0 = wmma_bf16(a, b0, acc0);
    acc1 = wmma_bf16(a, b1, acc1);
    __syncthreads();
  }
  store_tile(Cb, DH_, m0 + wr * 16, wc * 32,      acc0, 1.f, nullptr, MP_, DH_);
  store_tile(Cb, DH_, m0 + wr * 16, wc * 32 + 16, acc1, 1.f, nullptr, MP_, DH_);
}

// K8: d_inv[b,h,n] = 1 / (qf[n,:] . k_sum)
__global__ __launch_bounds__(256)
void dinv_kernel(const float* __restrict__ Qf, const float* __restrict__ ksum,
                 float* __restrict__ dinv) {
  const int lane = threadIdx.x & 31, wid = threadIdx.x >> 5;
  const int bh = blockIdx.y;
  const int n = blockIdx.x * 8 + wid;
  const float* qr = Qf + (size_t)bh * N_ * MP_ + (size_t)n * MP_;
  const float* ks = ksum + (size_t)bh * MP_;
  float s = 0.f;
  for (int m = lane; m < MP_; m += 32) s += qr[m] * ks[m];
  s = wave_sum(s);
  if (lane == 0) dinv[(size_t)bh * N_ + n] = 1.0f / s;
}

// ---------------------------------------------------------------------------
// K9: attn[b,n,h*64+d] = d_inv[n] * sum_m qf[n,m] * ctx[m,d]; K = MP_ = 9*32
// ---------------------------------------------------------------------------
__global__ __launch_bounds__(256)
void out_gemm_kernel(const float* __restrict__ Qf, const float* __restrict__ Ctx,
                     const float* __restrict__ dinv, float* __restrict__ Attn) {
  __shared__ unsigned AsW[64 * 20];       // qf tile: 64 x 32, stride 40
  __shared__ unsigned BsW[32 * 36];       // ctx tile: 32 x 64, stride 72
  const int tid = threadIdx.x;
  const int w = tid >> 5, wr = w >> 1, wc = w & 1;
  const int bh = blockIdx.z, b = bh >> 4, h = bh & 15;
  const int n0 = blockIdx.x * 64;
  const float* qfb  = Qf + (size_t)bh * N_ * MP_;
  const float* ctxb = Ctx + (size_t)bh * MP_ * DH_;
  const float* dv   = dinv + (size_t)bh * N_;
  float* Cb = Attn + (size_t)b * N_ * D_ + (size_t)h * DH_;

  const int ar  = tid >> 4;               // 0..15 (+16, 4 passes)
  const int ac2 = (tid & 15) * 2;
  const float* ag = qfb + (size_t)(n0 + ar) * MP_ + ac2;
  unsigned* asp = AsW + ar * 20 + (ac2 >> 1);

  const int brr = tid >> 5;               // 0..7 (+8, 4 passes)
  const int bc2 = (tid & 31) * 2;
  const float* bg = ctxb + (size_t)brr * DH_ + bc2;
  unsigned* bsp = BsW + brr * 36 + (bc2 >> 1);

  v8f acc0 = {}, acc1 = {};

  for (int k0 = 0; k0 < MP_; k0 += 32) {
#pragma unroll
    for (int p = 0; p < 4; ++p) {
      const float2 v = *(const float2*)(ag + (size_t)p * 16 * MP_ + k0);
      asp[p * 16 * 20] = pack2(v.x, v.y);
    }
#pragma unroll
    for (int p = 0; p < 4; ++p) {
      const float2 v = *(const float2*)(bg + (size_t)(k0 + p * 8) * DH_);
      bsp[p * 8 * 36] = pack2(v.x, v.y);
    }
    __syncthreads();
    const __bf16* As = (const __bf16*)AsW;
    const __bf16* Bp = (const __bf16*)BsW;
    const v16bf a  = frag_a(As, wr * 16, 40);
    const v16bf b0 = frag_b(Bp, wc * 32,      72);
    const v16bf b1 = frag_b(Bp, wc * 32 + 16, 72);
    acc0 = wmma_bf16(a, b0, acc0);
    acc1 = wmma_bf16(a, b1, acc1);
    __syncthreads();
  }
  store_tile(Cb, D_, n0 + wr * 16, wc * 32,      acc0, 1.f, dv, N_, 1 << 30);
  store_tile(Cb, D_, n0 + wr * 16, wc * 32 + 16, acc1, 1.f, dv, N_, 1 << 30);
}

// K11: y = x + proj_out + bo; LayerNorm(y) * gamma + beta. One block per row.
__global__ __launch_bounds__(256)
void ln_kernel(const float* __restrict__ X, const float* __restrict__ P,
               const float* __restrict__ bo, const float* __restrict__ gamma,
               const float* __restrict__ beta, float* __restrict__ Out) {
  __shared__ float sm[16];
  const int row = blockIdx.x;
  const int tid = threadIdx.x;
  const float4 xv = ((const float4*)(X + (size_t)row * D_))[tid];
  const float4 pv = ((const float4*)(P + (size_t)row * D_))[tid];
  const float4 bv = ((const float4*)bo)[tid];
  float4 y;
  y.x = xv.x + pv.x + bv.x;
  y.y = xv.y + pv.y + bv.y;
  y.z = xv.z + pv.z + bv.z;
  y.w = xv.w + pv.w + bv.w;
  float s  = y.x + y.y + y.z + y.w;
  float ss = y.x * y.x + y.y * y.y + y.z * y.z + y.w * y.w;
  s = wave_sum(s); ss = wave_sum(ss);
  if ((tid & 31) == 0) { sm[tid >> 5] = s; sm[8 + (tid >> 5)] = ss; }
  __syncthreads();
  float ts = 0.f, tss = 0.f;
#pragma unroll
  for (int k = 0; k < 8; ++k) { ts += sm[k]; tss += sm[8 + k]; }
  const float mu  = ts * (1.f / D_);
  const float var = tss * (1.f / D_) - mu * mu;
  const float rstd = rsqrtf(var + EPS_LN_);
  const float4 gv = ((const float4*)gamma)[tid];
  const float4 be = ((const float4*)beta)[tid];
  float4 o;
  o.x = (y.x - mu) * rstd * gv.x + be.x;
  o.y = (y.y - mu) * rstd * gv.y + be.y;
  o.z = (y.z - mu) * rstd * gv.z + be.z;
  o.w = (y.w - mu) * rstd * gv.w + be.w;
  ((float4*)(Out + (size_t)row * D_))[tid] = o;
}

// ---------------------------------------------------------------------------
extern "C" void kernel_launch(void* const* d_in, const int* in_sizes, int n_in,
                              void* d_out, int out_size, void* d_ws, size_t ws_size,
                              hipStream_t stream) {
  (void)in_sizes; (void)n_in; (void)out_size; (void)ws_size;
  const float* x     = (const float*)d_in[0];
  const float* Wq    = (const float*)d_in[1];
  const float* Wk    = (const float*)d_in[2];
  const float* Wv    = (const float*)d_in[3];
  const float* Wo    = (const float*)d_in[4];
  const float* bo    = (const float*)d_in[5];
  const float* proj  = (const float*)d_in[6];
  const float* gamma = (const float*)d_in[7];
  const float* beta  = (const float*)d_in[8];
  float* out = (float*)d_out;
  float* ws  = (float*)d_ws;

  // Workspace layout (floats). Total ~202.8M floats (~811 MB).
  const size_t SZ  = (size_t)B_ * N_ * D_;        // 16,777,216
  const size_t FSZ = (size_t)B_ * H_ * N_ * MP_;  // 75,497,472
  float* Qb   = ws;
  float* Kb   = Qb + SZ;
  float* Vb   = Kb + SZ;
  float* QF   = Vb + SZ;
  float* KF   = QF + FSZ;
  float* CTX  = KF + FSZ;                                  // [64,288,64]
  float* KSUM = CTX + (size_t)B_ * H_ * MP_ * DH_;         // [64,288]
  float* DINV = KSUM + (size_t)B_ * H_ * MP_;              // [64,4096]
  float* KMAX = DINV + (size_t)B_ * H_ * N_;               // [64]
  float* ATTN = Qb;   // Q dead after qexp
  float* PRJ  = Kb;   // K dead after kexp

  const dim3 gProj(B_ * N_ / 128, D_ / 128);      // (128,8)
  gemm_xw_kernel<<<gProj, 256, 0, stream>>>(x, Wq, Qb);
  gemm_xw_kernel<<<gProj, 256, 0, stream>>>(x, Wk, Kb);
  gemm_xw_kernel<<<gProj, 256, 0, stream>>>(x, Wv, Vb);

  const dim3 gFeat(N_ / 64, (MP_ + 63) / 64, B_ * H_);  // (64,5,64)
  feature_gemm_kernel<<<gFeat, 256, 0, stream>>>(Qb, proj, QF);
  feature_gemm_kernel<<<gFeat, 256, 0, stream>>>(Kb, proj, KF);

  const dim3 gRow(N_ / 8, B_ * H_);               // (512,64), 1 wave/row
  qexp_kernel<<<gRow, 256, 0, stream>>>(QF, Qb);
  kmax_kernel<<<B_ * H_, 1024, 0, stream>>>(KF, KMAX);
  kexp_kernel<<<gRow, 256, 0, stream>>>(KF, Kb, KMAX);
  ksum_kernel<<<B_ * H_, MP_, 0, stream>>>(KF, KSUM);

  const dim3 gCtx((MP_ + 63) / 64, 1, B_ * H_);   // (5,1,64)
  ctx_gemm_kernel<<<gCtx, 256, 0, stream>>>(KF, Vb, CTX);

  dinv_kernel<<<gRow, 256, 0, stream>>>(QF, KSUM, DINV);

  const dim3 gOut(N_ / 64, 1, B_ * H_);           // (64,1,64)
  out_gemm_kernel<<<gOut, 256, 0, stream>>>(QF, CTX, DINV, ATTN);

  gemm_xw_kernel<<<gProj, 256, 0, stream>>>(ATTN, Wo, PRJ);
  ln_kernel<<<B_ * N_, 256, 0, stream>>>(x, PRJ, bo, gamma, beta, out);
}
